// BahdanauAttention_5196910428839
// MI455X (gfx1250) — matrix operand order repaired and checked by reading, not compiled, not run on hardware
//
#include <hip/hip_runtime.h>
#include <hip/hip_bf16.h>

#define BB 128
#define SS 2048
#define HH 512

typedef __bf16 bf16_t;
typedef __attribute__((ext_vector_type(16))) __bf16 v16bf;
typedef __attribute__((ext_vector_type(8)))  __bf16 v8bf;
typedef __attribute__((ext_vector_type(8)))  float  v8f;
typedef __attribute__((ext_vector_type(4)))  int    v4i;

typedef __attribute__((address_space(1))) v4i* as1_v4i_p;  // global int4*
typedef __attribute__((address_space(3))) v4i* as3_v4i_p;  // LDS int4*

// ---------- CDNA5 async-to-LDS helpers (builtin if present, else inline asm) ----------
__device__ __forceinline__ void async_ld_b128(const bf16_t* g, bf16_t* l) {
#if __has_builtin(__builtin_amdgcn_global_load_async_to_lds_b128)
    __builtin_amdgcn_global_load_async_to_lds_b128(
        (as1_v4i_p)g, (as3_v4i_p)l, 0, 0);
#else
    unsigned lofs = (unsigned)(unsigned long long)l;   // LDS addr = low 32 bits
    asm volatile("global_load_async_to_lds_b128 %0, %1, off"
                 :: "v"(lofs), "v"(g) : "memory");
#endif
}

template <int N>
__device__ __forceinline__ void wait_asynccnt() {
#if __has_builtin(__builtin_amdgcn_s_wait_asynccnt)
    __builtin_amdgcn_s_wait_asynccnt(N);
#else
    if constexpr (N == 0)      asm volatile("s_wait_asynccnt 0x0" ::: "memory");
    else if constexpr (N == 2) asm volatile("s_wait_asynccnt 0x2" ::: "memory");
    else                       asm volatile("s_wait_asynccnt 0x3f" ::: "memory");
#endif
}

// ---------------- Kernel 0a: wq[b,o] = sum_h query[b,h]*Wa[o,h] + ba[o] ----
__global__ __launch_bounds__(256)
void wq_kernel(const float* __restrict__ query, const float* __restrict__ Wa,
               const float* __restrict__ ba, float* __restrict__ wq) {
    int b    = blockIdx.y;
    int wv   = threadIdx.x >> 5;
    int lane = threadIdx.x & 31;
    int o    = blockIdx.x * 8 + wv;
    const float* q    = query + (size_t)b * HH;
    const float* wrow = Wa + (size_t)o * HH;
    float acc = 0.f;
    for (int h = lane; h < HH; h += 32) acc += q[h] * wrow[h];
    #pragma unroll
    for (int off = 16; off > 0; off >>= 1) acc += __shfl_down(acc, off);
    if (lane == 0) wq[(size_t)b * HH + o] = acc + ba[o];
}

// ---------------- Kernel 0b: Ua f32 -> bf16 --------------------------------
__global__ __launch_bounds__(256)
void ua_cvt_kernel(const float* __restrict__ Ua, bf16_t* __restrict__ out) {
    int idx = (blockIdx.x * 256 + threadIdx.x) * 4;
    float4 v = *(const float4*)(Ua + idx);
    out[idx + 0] = (bf16_t)v.x;
    out[idx + 1] = (bf16_t)v.y;
    out[idx + 2] = (bf16_t)v.z;
    out[idx + 3] = (bf16_t)v.w;
}

// ---------------- Kernel 1: scores via WMMA bf16 + async B staging ---------
// One WG = one batch b, one 32-row tile of S.  score[s] = Va . tanh(wq+bu+uk[s,:]) (+bv)
// Wave wv owns a 64-col N slice and BOTH 16-row M subtiles (B register reuse x2).
#define STILE 32
#define LSTR  520   // bf16 elems per LDS row (512 + 8 pad: bank spread, keeps 16B align)

__global__ __launch_bounds__(256)
void score_kernel(const float* __restrict__ keys, const float* __restrict__ wq,
                  const bf16_t* __restrict__ ua, const float* __restrict__ bu,
                  const float* __restrict__ Va, const float* __restrict__ bv,
                  float* __restrict__ scores) {
    __shared__ __align__(16) bf16_t tile[STILE * LSTR];      // keys tile, bf16
    __shared__ __align__(16) bf16_t bstage[8][2][16 * 32];   // per-wave double-buffered B panel
    __shared__ float sPartial[STILE];

    const int b   = blockIdx.y;
    const int s0  = blockIdx.x * STILE;
    const int tid = threadIdx.x;

    if (tid < STILE) sPartial[tid] = 0.f;

    // Cooperative load of keys[b, s0:s0+32, :] with f32->bf16 convert.
    const float* kbase = keys + ((size_t)b * SS + s0) * HH;
    #pragma unroll 4
    for (int i = 0; i < 16; ++i) {
        int e   = (tid + i * 256) * 4;      // flat float index == row*512 + col
        int row = e >> 9;
        int col = e & 511;
        float4 v = *(const float4*)(kbase + e);
        bf16_t* d = &tile[row * LSTR + col];
        d[0] = (bf16_t)v.x; d[1] = (bf16_t)v.y;
        d[2] = (bf16_t)v.z; d[3] = (bf16_t)v.w;
    }
    __syncthreads();

    const int wv     = tid >> 5;
    const int lane   = tid & 31;
    const int laneLo = lane & 15;
    const int laneHi = lane >> 4;       // 0: lanes 0-15, 1: lanes 16-31

    // A fragments (16-bit A 16x32 layout): lane m holds K {0..7,16..23} (lo half)
    // or K {8..15,24..31} (hi half) of row m.  Two M subtiles per wave.
    const bf16_t* aRow0 = &tile[(laneLo)      * LSTR + laneHi * 8];
    const bf16_t* aRow1 = &tile[(16 + laneLo) * LSTR + laneHi * 8];

    // async B staging: lane (col = lane&15, khalf = lane>>4) fetches two 16B chunks.
    // LDS panel layout: [col][k], stride 32 bf16 (64B per col).
    const int sCol = laneLo;
    const int sKof = laneHi * 8;

    float racc0[8], racc1[8];
    #pragma unroll
    for (int r = 0; r < 8; ++r) { racc0[r] = 0.f; racc1[r] = 0.f; }

    for (int nt = 0; nt < 4; ++nt) {
        const int nBase = wv * 64 + nt * 16;
        const int ocol  = nBase + laneLo;

        // prologue: stage B(k0=0) into buffer 0
        {
            const bf16_t* g = ua + (size_t)(nBase + sCol) * HH + sKof;
            bf16_t*       l = &bstage[wv][0][sCol * 32 + sKof];
            async_ld_b128(g, l);            // K [k0, k0+16)
            async_ld_b128(g + 16, l + 16);  // K [k0+16, k0+32)
        }

        v8f c0 = {0.f,0.f,0.f,0.f,0.f,0.f,0.f,0.f};
        v8f c1 = {0.f,0.f,0.f,0.f,0.f,0.f,0.f,0.f};

        #pragma unroll
        for (int k0 = 0; k0 < HH; k0 += 32) {
            const int curb = (k0 >> 5) & 1;
            if (k0 + 32 < HH) {
                // stage next panel into the other buffer, then wait only for current
                const bf16_t* g = ua + (size_t)(nBase + sCol) * HH + (k0 + 32) + sKof;
                bf16_t*       l = &bstage[wv][curb ^ 1][sCol * 32 + sKof];
                async_ld_b128(g, l);
                async_ld_b128(g + 16, l + 16);
                wait_asynccnt<2>();         // in-order: current panel's 2 loads done
            } else {
                wait_asynccnt<0>();         // tail: drain
            }

            // B fragment: lanes 0-15 K {0..7,16..23} of col, lanes 16-31 K {8..15,24..31}
            const bf16_t* bb = &bstage[wv][curb][laneLo * 32 + laneHi * 8];
            v8bf blo = *(const v8bf*)(bb);
            v8bf bhi = *(const v8bf*)(bb + 16);
            v8bf a0lo = *(const v8bf*)(aRow0 + k0);
            v8bf a0hi = *(const v8bf*)(aRow0 + k0 + 16);
            v8bf a1lo = *(const v8bf*)(aRow1 + k0);
            v8bf a1hi = *(const v8bf*)(aRow1 + k0 + 16);
            v16bf a0, a1, bm;
            #pragma unroll
            for (int i = 0; i < 8; ++i) {
                a0[i] = a0lo[i]; a0[i + 8] = a0hi[i];
                a1[i] = a1lo[i]; a1[i + 8] = a1hi[i];
                bm[i] = blo[i];  bm[i + 8] = bhi[i];
            }
            c0 = __builtin_amdgcn_wmma_f32_16x16x32_bf16(
                     false, a0, false, bm, (short)0, c0, false, false);
            c1 = __builtin_amdgcn_wmma_f32_16x16x32_bf16(
                     false, a1, false, bm, (short)0, c1, false, false);
        }

        // c0[r] = uk(row r + laneHi*8, ocol); c1[r] = uk(row 16 + r + laneHi*8, ocol)
        const float colAdd = wq[(size_t)b * HH + ocol] + bu[ocol];
        const float vav    = Va[ocol];
        #pragma unroll
        for (int r = 0; r < 8; ++r) {
            racc0[r] += tanhf(c0[r] + colAdd) * vav;
            racc1[r] += tanhf(c1[r] + colAdd) * vav;
        }
    }

    #pragma unroll
    for (int r = 0; r < 8; ++r) {
        atomicAdd(&sPartial[r + laneHi * 8],      racc0[r]);   // ds_add_f32
        atomicAdd(&sPartial[16 + r + laneHi * 8], racc1[r]);
    }
    __syncthreads();

    if (tid < STILE)
        scores[(size_t)b * SS + s0 + tid] = sPartial[tid] + bv[0];
}

// ---------------- Kernel 2: softmax over S + context = weights . keys ------
__global__ __launch_bounds__(256)
void softmax_ctx_kernel(const float* __restrict__ keys, float* __restrict__ out) {
    __shared__ float sw[SS];
    __shared__ float red[256];
    const int b   = blockIdx.x;
    const int tid = threadIdx.x;

    const float* sc = out + (size_t)BB * HH + (size_t)b * SS;  // raw scores (weights region)

    float lmax = -3.4e38f;
    #pragma unroll
    for (int i = 0; i < SS / 256; ++i) {
        int idx = tid + i * 256;
        float v = sc[idx];
        sw[idx] = v;
        lmax = fmaxf(lmax, v);
    }
    red[tid] = lmax; __syncthreads();
    for (int off = 128; off > 0; off >>= 1) {
        if (tid < off) red[tid] = fmaxf(red[tid], red[tid + off]);
        __syncthreads();
    }
    const float m = red[0];
    __syncthreads();

    float lsum = 0.f;
    #pragma unroll
    for (int i = 0; i < SS / 256; ++i) {
        int idx = tid + i * 256;
        float e = __expf(sw[idx] - m);
        sw[idx] = e;
        lsum += e;
    }
    red[tid] = lsum; __syncthreads();
    for (int off = 128; off > 0; off >>= 1) {
        if (tid < off) red[tid] += red[tid + off];
        __syncthreads();
    }
    const float inv = 1.f / red[0];

    float* wout = out + (size_t)BB * HH + (size_t)b * SS;
    #pragma unroll
    for (int i = 0; i < SS / 256; ++i) {
        int idx = tid + i * 256;
        wout[idx] = sw[idx] * inv;
    }

    // context[b,h] = inv * sum_s e[s] * keys[b,s,h]; 2 columns per thread (coalesced)
    const float* kb = keys + (size_t)b * SS * HH;
    const int h0 = tid, h1 = tid + 256;
    float acc0 = 0.f, acc1 = 0.f;
    #pragma unroll 4
    for (int s = 0; s < SS; ++s) {
        float w = sw[s];
        acc0 += w * kb[(size_t)s * HH + h0];
        acc1 += w * kb[(size_t)s * HH + h1];
    }
    out[(size_t)b * HH + h0] = acc0 * inv;
    out[(size_t)b * HH + h1] = acc1 * inv;
}

// ---------------- launch ---------------------------------------------------
extern "C" void kernel_launch(void* const* d_in, const int* in_sizes, int n_in,
                              void* d_out, int out_size, void* d_ws, size_t ws_size,
                              hipStream_t stream) {
    (void)in_sizes; (void)n_in; (void)out_size; (void)ws_size;
    const float* query = (const float*)d_in[0];
    const float* keys  = (const float*)d_in[1];
    const float* Wa    = (const float*)d_in[2];
    const float* ba    = (const float*)d_in[3];
    const float* Ua    = (const float*)d_in[4];
    const float* bu    = (const float*)d_in[5];
    const float* Va    = (const float*)d_in[6];
    const float* bv    = (const float*)d_in[7];

    float*  out    = (float*)d_out;
    char*   ws     = (char*)d_ws;
    float*  wq     = (float*)ws;                         // 128*512*4 = 256 KB
    bf16_t* ua16   = (bf16_t*)(ws + BB * HH * 4);        // 512*512*2 = 512 KB
    float*  scores = out + (size_t)BB * HH;              // raw scores in weights region

    wq_kernel<<<dim3(HH / 8, BB), 256, 0, stream>>>(query, Wa, ba, wq);
    ua_cvt_kernel<<<(HH * HH) / (256 * 4), 256, 0, stream>>>(Ua, ua16);
    score_kernel<<<dim3(SS / STILE, BB), 256, 0, stream>>>(keys, wq, ua16, bu, Va, bv, scores);
    softmax_ctx_kernel<<<BB, 256, 0, stream>>>(keys, out);
}
